// PBRNN_22505628631359
// MI455X (gfx1250) — matrix operand order repaired and checked by reading, not compiled, not run on hardware
//
#include <hip/hip_runtime.h>
#include <hip/hip_bf16.h>

// Problem dims (match reference)
#define BDIM 64
#define SEQ  512
#define DIN  544
#define HID  1024
#define NBLK 64          // HID/16 persistent workgroups
#define NTHREADS 128     // 4 waves (wave32)

typedef __attribute__((ext_vector_type(16))) __bf16 v16bf;
typedef __attribute__((ext_vector_type(8)))  __bf16 v8bf;
typedef __attribute__((ext_vector_type(8)))  float  v8f;
typedef __attribute__((ext_vector_type(4)))  unsigned tdm4;
typedef __attribute__((ext_vector_type(8)))  unsigned tdm8;

union frag16 { v16bf v; v8bf h[2]; };

// ---------------- prep kernels ----------------

// X [B,S,D] fp32 -> xbf [S,B,D] bf16 (seq-major so each step reads a dense 64x544 tile)
__global__ void pbrnn_prep_x(const float* __restrict__ X, __bf16* __restrict__ xbf) {
  size_t i = (size_t)blockIdx.x * blockDim.x + threadIdx.x;
  const size_t total = (size_t)SEQ * BDIM * DIN;
  if (i >= total) return;
  int d  = (int)(i % DIN);
  size_t sb = i / DIN;
  int b  = (int)(sb % BDIM);
  int s  = (int)(sb / BDIM);
  xbf[i] = (__bf16)X[((size_t)b * SEQ + s) * DIN + d];
}

// W_ih, W_hh -> bf16 copies; b4 = b_ih + b_hh
__global__ void pbrnn_prep_w(const float* __restrict__ Wih, const float* __restrict__ Whh,
                             const float* __restrict__ bih, const float* __restrict__ bhh,
                             __bf16* __restrict__ wihb, __bf16* __restrict__ whhb,
                             float* __restrict__ b4) {
  size_t i = (size_t)blockIdx.x * blockDim.x + threadIdx.x;
  const size_t NW1 = (size_t)4 * HID * DIN;
  const size_t NW2 = (size_t)4 * HID * HID;
  if (i < NW1) {
    wihb[i] = (__bf16)Wih[i];
  } else if (i < NW1 + NW2) {
    size_t j = i - NW1;
    whhb[j] = (__bf16)Whh[j];
  } else if (i < NW1 + NW2 + (size_t)4 * HID) {
    size_t j = i - NW1 - NW2;
    b4[j] = bih[j] + bhh[j];
  }
}

// zero h double-buffer and the grid-sync counter (re-run every launch -> deterministic)
__global__ void pbrnn_zero(__bf16* __restrict__ hbuf, unsigned* __restrict__ cnt) {
  int i = blockIdx.x * blockDim.x + threadIdx.x;
  if (i < 2 * BDIM * HID) hbuf[i] = (__bf16)0.f;
  if (i == 0) *cnt = 0u;
}

// ---------------- activations ----------------

__device__ __forceinline__ float sigmoid_(float x) {
  return 1.f / (1.f + __expf(-x));
}
__device__ __forceinline__ float tanh_(float x) {
  x = fminf(15.f, fmaxf(-15.f, x));
  float e = __expf(2.f * x);
  return (e - 1.f) / (e + 1.f);
}

// ---------------- persistent LSTM kernel ----------------
// grid = 64 blocks (one per 16-wide hidden slice), 128 threads (4 waves).
// Wave w computes batch rows [16w, 16w+16) for all 4 gates of this slice.
// LDS holds this block's weight slice: 4 gates x 16 rows x (DIN + HID) bf16 = 196 KB,
// staged once via TDM (W_ih tiles) + async global->LDS (W_hh tiles).

extern __shared__ __bf16 pbrnn_smem[];

__global__ __launch_bounds__(NTHREADS, 1)
void pbrnn_lstm(const __bf16* __restrict__ xbf,
                const __bf16* __restrict__ wihb,
                const __bf16* __restrict__ whhb,
                const float*  __restrict__ b4,
                __bf16*       __restrict__ hbuf,   // [2][B][H] bf16 double buffer
                unsigned*     __restrict__ cnt,
                float*        __restrict__ out) {  // [2][B][H] fp32 (h_n, c_n)
  const int tid  = threadIdx.x;
  const int lane = tid & 31;
  const int wv   = tid >> 5;           // 0..3 -> M tile
  const int j0   = blockIdx.x * 16;    // hidden-column slice

  __bf16* wih_s = pbrnn_smem;                       // [4][16][DIN]
  __bf16* whh_s = pbrnn_smem + 4 * 16 * DIN;        // [4][16][HID]

  // --- stage W_ih gate tiles via the Tensor Data Mover (one 2D D# per gate) ---
  // Tile: 16 rows x DIN bf16, row stride DIN elements, packed contiguously into LDS.
  if (wv == 0) {
#pragma unroll
    for (int g = 0; g < 4; ++g) {
      unsigned lds_addr = (unsigned)(uintptr_t)(wih_s + (size_t)g * 16 * DIN);
      unsigned long long ga =
          (unsigned long long)(uintptr_t)(wihb + (size_t)(g * HID + j0) * DIN);
      tdm4 g0; tdm8 g1; tdm4 gz;
      gz[0] = 0u; gz[1] = 0u; gz[2] = 0u; gz[3] = 0u;
      g0[0] = 1u;                                        // count=1, user mode
      g0[1] = lds_addr;                                  // LDS byte address
      g0[2] = (unsigned)(ga & 0xFFFFFFFFu);              // global_addr[31:0]
      g0[3] = (unsigned)((ga >> 32) & 0x01FFFFFFu)       // global_addr[56:32]
              | (2u << 30);                              // type=2 (image)
      g1[0] = 1u << 16;                                  // data_size=1 (2 bytes)
      g1[1] = ((unsigned)DIN & 0xFFFFu) << 16;           // tensor_dim0[15:0]
      g1[2] = 16u << 16;                                 // tensor_dim1[15:0]
      g1[3] = ((unsigned)DIN) << 16;                     // tile_dim0
      g1[4] = 16u;                                       // tile_dim1 (tile_dim2=0)
      g1[5] = (unsigned)DIN;                             // tensor_dim0_stride[31:0]
      g1[6] = 0u;
      g1[7] = 0u;
      asm volatile("tensor_load_to_lds %0, %1, %2, %3"
                   :: "s"(g0), "s"(g1), "s"(gz), "s"(gz)
                   : "memory");
    }
  }

  // --- stage W_hh gate tiles via async global->LDS (per-lane 16B transfers) ---
  for (int idx = tid; idx < 4 * 16 * (HID / 8); idx += NTHREADS) {
    int v  = idx % (HID / 8);
    int rn = idx / (HID / 8);
    int g = rn >> 4, n = rn & 15;
    unsigned ldsoff =
        (unsigned)(uintptr_t)(whh_s + (size_t)(g * 16 + n) * HID + v * 8);
    unsigned long long gaddr = (unsigned long long)(uintptr_t)(
        whhb + ((size_t)(g * HID + j0 + n)) * HID + v * 8);
    asm volatile("global_load_async_to_lds_b128 %0, %1, off"
                 :: "v"(ldsoff), "v"(gaddr)
                 : "memory");
  }
  asm volatile("s_wait_asynccnt 0x0" ::: "memory");
  if (wv == 0) __builtin_amdgcn_s_wait_tensorcnt(0);

  // per-lane gate biases (C/D layout: N column = lane & 15 for both lane halves)
  float bias[4];
#pragma unroll
  for (int g = 0; g < 4; ++g) bias[g] = b4[g * HID + j0 + (lane & 15)];

  __syncthreads();

  // WMMA fragment lane geometry (16-bit A 16x32 and B 32x16 layouts)
  const int mrow = wv * 16;
  const int am   = lane & 15;              // A: M = lane&15
  const int aK0  = (lane >> 4) ? 8 : 0;    // A: K offset of this lane-half
  const int bn   = lane & 15;              // B: N = lane&15
  const int bK0  = (lane >> 4) ? 8 : 0;    // B: K offset of this lane-half

  v8f zero8 = {};
  v8f acc_c = zero8;                       // cell state, lives in registers
  float hfin[8];
#pragma unroll
  for (int r = 0; r < 8; ++r) hfin[r] = 0.f;

  for (int s = 0; s < SEQ; ++s) {
    v8f acc[4];
#pragma unroll
    for (int g = 0; g < 4; ++g) acc[g] = zero8;

    // ---- input projection contribution: gates += x_s @ W_ih^T ----
    const __bf16* xrow = xbf + ((size_t)s * BDIM + mrow + am) * DIN;
#pragma unroll 2
    for (int k = 0; k < DIN; k += 32) {
      frag16 a;
      a.h[0] = *(const v8bf*)(xrow + k + aK0);
      a.h[1] = *(const v8bf*)(xrow + k + 16 + aK0);
#pragma unroll
      for (int g = 0; g < 4; ++g) {
        frag16 bfr;
        const __bf16* wp = wih_s + (size_t)(g * 16 + bn) * DIN + k + bK0;
        bfr.h[0] = *(const v8bf*)(wp);
        bfr.h[1] = *(const v8bf*)(wp + 16);
        acc[g] = __builtin_amdgcn_wmma_f32_16x16x32_bf16(
            false, a.v, false, bfr.v, (short)0, acc[g], false, false);
      }
    }

    // ---- recurrent contribution: gates += h @ W_hh^T ----
    const __bf16* hrow = hbuf + ((size_t)(s & 1) * BDIM + mrow + am) * HID;
#pragma unroll 2
    for (int k = 0; k < HID; k += 32) {
      frag16 a;
      a.h[0] = *(const v8bf*)(hrow + k + aK0);
      a.h[1] = *(const v8bf*)(hrow + k + 16 + aK0);
#pragma unroll
      for (int g = 0; g < 4; ++g) {
        frag16 bfr;
        const __bf16* wp = whh_s + (size_t)(g * 16 + bn) * HID + k + bK0;
        bfr.h[0] = *(const v8bf*)(wp);
        bfr.h[1] = *(const v8bf*)(wp + 16);
        acc[g] = __builtin_amdgcn_wmma_f32_16x16x32_bf16(
            false, a.v, false, bfr.v, (short)0, acc[g], false, false);
      }
    }

    // prefetch next step's x rows while we do the element-wise cell update
    if (s + 1 < SEQ) {
      const __bf16* xn = xbf + ((size_t)(s + 1) * BDIM + mrow + am) * DIN;
      __builtin_prefetch(xn, 0, 1);
    }

    // ---- LSTM cell update, purely per-lane in the C/D accumulator layout ----
    __bf16* hdst = hbuf + (size_t)((s + 1) & 1) * BDIM * HID;
    const int rowadd = mrow + ((lane >> 4) << 3);   // M = r + 8*(lane>=16)
    const int col    = j0 + (lane & 15);
#pragma unroll
    for (int r = 0; r < 8; ++r) {
      float iv = sigmoid_(acc[0][r] + bias[0]);
      float fv = sigmoid_(acc[1][r] + bias[1]);
      float gv = tanh_(acc[2][r] + bias[2]);
      float ov = sigmoid_(acc[3][r] + bias[3]);
      float cv = fv * acc_c[r] + iv * gv;
      acc_c[r] = cv;
      float hv = ov * tanh_(cv);
      hfin[r]  = hv;
      hdst[(size_t)(rowadd + r) * HID + col] = (__bf16)hv;
    }

    // ---- grid-wide step barrier (device-scope atomic, double-buffered h) ----
    if (s + 1 < SEQ) {
      __threadfence();
      __syncthreads();
      if (tid == 0) {
        __hip_atomic_fetch_add(cnt, 1u, __ATOMIC_ACQ_REL, __HIP_MEMORY_SCOPE_AGENT);
        const unsigned target = (unsigned)NBLK * (unsigned)(s + 1);
        while (__hip_atomic_load(cnt, __ATOMIC_ACQUIRE, __HIP_MEMORY_SCOPE_AGENT) < target) {
          __builtin_amdgcn_s_sleep(2);
        }
      }
      __syncthreads();
    }
  }

  // ---- final state: out = stack([h_n, c_n]) as [2, B, H] fp32 ----
  const int rowadd = mrow + ((lane >> 4) << 3);
  const int col    = j0 + (lane & 15);
#pragma unroll
  for (int r = 0; r < 8; ++r) {
    out[(size_t)(rowadd + r) * HID + col] = hfin[r];
    out[(size_t)BDIM * HID + (size_t)(rowadd + r) * HID + col] = acc_c[r];
  }
}

// ---------------- launch ----------------

extern "C" void kernel_launch(void* const* d_in, const int* in_sizes, int n_in,
                              void* d_out, int out_size, void* d_ws, size_t ws_size,
                              hipStream_t stream) {
  (void)in_sizes; (void)n_in; (void)out_size; (void)ws_size;

  const float* X    = (const float*)d_in[0];   // [B,S,D]
  const float* W_ih = (const float*)d_in[1];   // [4H,D]
  const float* W_hh = (const float*)d_in[2];   // [4H,H]
  const float* b_ih = (const float*)d_in[3];   // [4H]
  const float* b_hh = (const float*)d_in[4];   // [4H]
  float* out = (float*)d_out;                  // [2,B,H]

  // workspace layout (256B aligned sections)
  char* ws = (char*)d_ws;
  constexpr size_t SZ_XBF = (size_t)SEQ * BDIM * DIN * 2;        // 35,651,584
  constexpr size_t SZ_WIH = (size_t)4 * HID * DIN * 2;           //  4,456,448
  constexpr size_t SZ_WHH = (size_t)4 * HID * HID * 2;           //  8,388,608
  constexpr size_t SZ_B4  = (size_t)4 * HID * 4;                 //     16,384
  constexpr size_t SZ_H   = (size_t)2 * BDIM * HID * 2;          //    262,144
  __bf16*   xbf  = (__bf16*)(ws);
  __bf16*   wihb = (__bf16*)(ws + SZ_XBF);
  __bf16*   whhb = (__bf16*)(ws + SZ_XBF + SZ_WIH);
  float*    b4   = (float*) (ws + SZ_XBF + SZ_WIH + SZ_WHH);
  __bf16*   hbuf = (__bf16*)(ws + SZ_XBF + SZ_WIH + SZ_WHH + SZ_B4);
  unsigned* cnt  = (unsigned*)(ws + SZ_XBF + SZ_WIH + SZ_WHH + SZ_B4 + SZ_H);

  // 1) convert X to bf16 [S,B,D]
  {
    const size_t total = (size_t)SEQ * BDIM * DIN;
    dim3 grid((unsigned)((total + 255) / 256));
    pbrnn_prep_x<<<grid, 256, 0, stream>>>(X, xbf);
  }
  // 2) convert weights to bf16, fuse biases
  {
    const size_t total = (size_t)4 * HID * DIN + (size_t)4 * HID * HID + (size_t)4 * HID;
    dim3 grid((unsigned)((total + 255) / 256));
    pbrnn_prep_w<<<grid, 256, 0, stream>>>(W_ih, W_hh, b_ih, b_hh, wihb, whhb, b4);
  }
  // 3) zero h double-buffer + sync counter
  {
    dim3 grid((2 * BDIM * HID + 255) / 256);
    pbrnn_zero<<<grid, 256, 0, stream>>>(hbuf, cnt);
  }
  // 4) persistent fused LSTM: 64 WGs x 128 threads, 196 KB dynamic LDS each
  {
    const size_t shmem = ((size_t)4 * 16 * DIN + (size_t)4 * 16 * HID) * 2;  // 200,704 B
    pbrnn_lstm<<<NBLK, NTHREADS, shmem, stream>>>(xbf, wihb, whhb, b4, hbuf, cnt, out);
  }
}
